// Attention_48773648613848
// MI455X (gfx1250) — matrix operand order repaired and checked by reading, not compiled, not run on hardware
//
#include <hip/hip_runtime.h>
#include <hip/hip_bf16.h>
#include <math.h>

typedef __attribute__((ext_vector_type(16))) _Float16 v16h;
typedef __attribute__((ext_vector_type(8)))  _Float16 v8h;
typedef __attribute__((ext_vector_type(8)))  float    v8f;
typedef __attribute__((ext_vector_type(4)))  float    v4f;

#define HEADS  12
#define DHEAD  64
#define DIM    768
#define INNER  768
#define BATCH  2
#define SEQ    2048
#define NROWS  (BATCH*SEQ)        // 4096
#define QKV3   (INNER*3)          // 2304
#define SCALE_F 0.125f            // 64^-0.5
#define NEG_F  (-1e9f)
#define EPS_F  (1e-5f)
#define FMIN_F (-3.0e38f)

// ---------------------------------------------------------------------------
// WMMA helper: D = A(16x32 f16) * B(32x16 f16) + C(16x16 f32)
// ---------------------------------------------------------------------------
__device__ __forceinline__ v8f wmma16(v16h a, v16h b, v8f c) {
  return __builtin_amdgcn_wmma_f32_16x16x32_f16(
      /*neg_a=*/false, a, /*neg_b=*/false, b,
      /*c_mod=*/(short)0, c, /*reuse_a=*/false, /*reuse_b=*/false);
}

// A fragment (16x32, 16-bit) from row-major storage, ld in halves.
// Lane l: row = l&15, k-halves [klo..klo+7] and [16+klo..16+klo+7], klo=(l>>4)*8
__device__ __forceinline__ v16h frag_a(const _Float16* base, int ld) {
  int l = threadIdx.x & 31;
  const _Float16* p = base + (size_t)(l & 15) * ld + ((l >> 4) << 3);
  v8h lo = *(const v8h*)p;
  v8h hi = *(const v8h*)(p + 16);
  v16h r;
#pragma unroll
  for (int i = 0; i < 8; ++i) { r[i] = lo[i]; r[i + 8] = hi[i]; }
  return r;
}

// B fragment (32x16, 16-bit) from B^T storage [N][K], ld in halves.
// Lane l: n = l&15, k-halves [klo..klo+7] and [16+klo..16+klo+7], klo=(l>>4)*8
__device__ __forceinline__ v16h frag_bt(const _Float16* base, int ld) {
  int l = threadIdx.x & 31;
  const _Float16* p = base + (size_t)(l & 15) * ld + ((l >> 4) << 3);
  v8h lo = *(const v8h*)p;
  v8h hi = *(const v8h*)(p + 16);
  v16h r;
#pragma unroll
  for (int i = 0; i < 8; ++i) { r[i] = lo[i]; r[i + 8] = hi[i]; }
  return r;
}

// ---------------------------------------------------------------------------
// 1) LayerNorm -> xn (f16), one block per row of 768
// ---------------------------------------------------------------------------
__global__ void ln_kernel(const float* __restrict__ x,
                          const float* __restrict__ gamma,
                          const float* __restrict__ beta,
                          _Float16* __restrict__ xn) {
  __shared__ float red[256];
  int row = blockIdx.x;
  int t = threadIdx.x;
  const float* xr = x + (size_t)row * DIM;
  float v0 = xr[t], v1 = xr[t + 256], v2 = xr[t + 512];
  red[t] = v0 + v1 + v2;
  __syncthreads();
  for (int off = 128; off > 0; off >>= 1) {
    if (t < off) red[t] += red[t + off];
    __syncthreads();
  }
  float mu = red[0] * (1.0f / DIM);
  __syncthreads();
  float d0 = v0 - mu, d1 = v1 - mu, d2 = v2 - mu;
  red[t] = d0 * d0 + d1 * d1 + d2 * d2;
  __syncthreads();
  for (int off = 128; off > 0; off >>= 1) {
    if (t < off) red[t] += red[t + off];
    __syncthreads();
  }
  float rstd = rsqrtf(red[0] * (1.0f / DIM) + EPS_F);
  _Float16* xo = xn + (size_t)row * DIM;
  xo[t]       = (_Float16)(d0 * rstd * gamma[t]       + beta[t]);
  xo[t + 256] = (_Float16)(d1 * rstd * gamma[t + 256] + beta[t + 256]);
  xo[t + 512] = (_Float16)(d2 * rstd * gamma[t + 512] + beta[t + 512]);
}

// ---------------------------------------------------------------------------
// 2) Weight transpose + f32->f16:  wT[c][r] = w[r][c]
// ---------------------------------------------------------------------------
__global__ void wt_kernel(const float* __restrict__ w, _Float16* __restrict__ wT,
                          int rows, int cols) {
  size_t i = (size_t)blockIdx.x * 256 + threadIdx.x;
  if (i < (size_t)rows * cols) {
    int r = (int)(i / cols), c = (int)(i % cols);
    wT[(size_t)c * rows + r] = (_Float16)w[i];
  }
}

// ---------------------------------------------------------------------------
// 3) QKV GEMM: [4096x768]f16 x [768x2304]f16 -> scatter into q,k (row-major
//    per head) and vT (transposed per head).  64x64 tile per block, 8 waves.
// ---------------------------------------------------------------------------
__global__ void qkv_gemm(const _Float16* __restrict__ xn,
                         const _Float16* __restrict__ wT,
                         _Float16* __restrict__ q,
                         _Float16* __restrict__ k,
                         _Float16* __restrict__ vT) {
  int wave = threadIdx.x >> 5;
  int wm = wave & 3, wn = wave >> 2;
  int row0 = blockIdx.y * 64 + wm * 16;
  int col0 = blockIdx.x * 64 + wn * 32;
  v8f acc0 = {}, acc1 = {};
  for (int kk = 0; kk < DIM; kk += 32) {
    v16h a  = frag_a(xn + (size_t)row0 * DIM + kk, DIM);
    v16h b0 = frag_bt(wT + (size_t)col0 * DIM + kk, DIM);
    v16h b1 = frag_bt(wT + (size_t)(col0 + 16) * DIM + kk, DIM);
    acc0 = wmma16(a, b0, acc0);
    acc1 = wmma16(a, b1, acc1);
  }
  int l = threadIdx.x & 31;
  int grp = l >> 4, ncol = l & 15;
#pragma unroll
  for (int t = 0; t < 2; ++t) {
    v8f acc = t ? acc1 : acc0;
    int c = col0 + t * 16 + ncol;
#pragma unroll
    for (int r = 0; r < 8; ++r) {
      int row = row0 + grp * 8 + r;
      int b_ = row >> 11, nn = row & (SEQ - 1);
      _Float16 h = (_Float16)acc[r];
      if (c < INNER) {
        int hh = c / DHEAD, dd = c % DHEAD;
        q[(((size_t)b_ * HEADS + hh) * SEQ + nn) * DHEAD + dd] = h;
      } else if (c < 2 * INNER) {
        int cc = c - INNER, hh = cc / DHEAD, dd = cc % DHEAD;
        k[(((size_t)b_ * HEADS + hh) * SEQ + nn) * DHEAD + dd] = h;
      } else {
        int cc = c - 2 * INNER, hh = cc / DHEAD, dd = cc % DHEAD;
        vT[(((size_t)b_ * HEADS + hh) * DHEAD + dd) * SEQ + nn] = h;
      }
    }
  }
}

// ---------------------------------------------------------------------------
// 4) S pass: S = (mask(Q K^T * scale)) * (1+con), streamed over keys with
//    ONLINE softmax stats (running max + rescaled running sum).
//    Block = 64 q-rows x all 2048 keys for one (b,h).  S written to attn buf.
// ---------------------------------------------------------------------------
__global__ void s_kernel(const _Float16* __restrict__ q,
                         const _Float16* __restrict__ k,
                         const long long* __restrict__ labels,
                         const float* __restrict__ con,
                         float* __restrict__ S,
                         float* __restrict__ rowmax,
                         float* __restrict__ rowsum) {
  __shared__ float smax[8][16];
  __shared__ float ssum[8][16];
  int bh = blockIdx.y;
  int b_ = bh / HEADS;
  int wave = threadIdx.x >> 5;
  int wm = wave & 3, wn = wave >> 2;
  int q0 = blockIdx.x * 64 + wm * 16;
  int l = threadIdx.x & 31;
  int grp = l >> 4, mo = grp * 8, ncol = l & 15;

  const _Float16* qb = q + (size_t)bh * SEQ * DHEAD;
  const _Float16* kb = k + (size_t)bh * SEQ * DHEAD;
  // Q fragments for both 32-wide k-chunks of d=64 (reused all iterations)
  v16h a0 = frag_a(qb + (size_t)q0 * DHEAD + 0,  DHEAD);
  v16h a1 = frag_a(qb + (size_t)q0 * DHEAD + 32, DHEAD);

  float m[8], s[8];
#pragma unroll
  for (int r = 0; r < 8; ++r) { m[r] = FMIN_F; s[r] = 0.0f; }

  for (int k0 = wn * 32; k0 < SEQ; k0 += 64) {
    v16h b0a = frag_bt(kb + (size_t)k0 * DHEAD + 0,         DHEAD);
    v16h b0b = frag_bt(kb + (size_t)k0 * DHEAD + 32,        DHEAD);
    v16h b1a = frag_bt(kb + (size_t)(k0 + 16) * DHEAD + 0,  DHEAD);
    v16h b1b = frag_bt(kb + (size_t)(k0 + 16) * DHEAD + 32, DHEAD);
    v8f acc0 = {}, acc1 = {};
    acc0 = wmma16(a0, b0a, acc0);
    acc0 = wmma16(a1, b0b, acc0);
    acc1 = wmma16(a0, b1a, acc1);
    acc1 = wmma16(a1, b1b, acc1);
    // prefetch next con block for this lane's first row
    __builtin_prefetch(con + ((size_t)b_ * SEQ + q0 + mo) * SEQ + k0 + 64, 0, 0);
#pragma unroll
    for (int t = 0; t < 2; ++t) {
      v8f acc = t ? acc1 : acc0;
      int key = k0 + t * 16 + ncol;
      bool pad = (labels[(size_t)b_ * SEQ + key] == 0);
#pragma unroll
      for (int r = 0; r < 8; ++r) {
        int qi = q0 + mo + r;
        float sv = acc[r] * SCALE_F;
        if (pad) sv = NEG_F;
        sv *= (1.0f + con[((size_t)b_ * SEQ + qi) * SEQ + key]);
        // online max/sum update
        float nm = fmaxf(m[r], sv);
        s[r] = s[r] * __expf(m[r] - nm) + __expf(sv - nm);
        m[r] = nm;
        S[((size_t)bh * SEQ + qi) * SEQ + key] = sv;
      }
    }
  }
  // butterfly reduce (m,s) across the 16 lanes holding one row
#pragma unroll
  for (int off = 1; off < 16; off <<= 1) {
#pragma unroll
    for (int r = 0; r < 8; ++r) {
      float om = __shfl_xor(m[r], off, 32);
      float os = __shfl_xor(s[r], off, 32);
      float nm = fmaxf(m[r], om);
      s[r] = s[r] * __expf(m[r] - nm) + os * __expf(om - nm);
      m[r] = nm;
    }
  }
  if ((l & 15) == 0) {
#pragma unroll
    for (int r = 0; r < 8; ++r) { smax[wave][mo + r] = m[r]; ssum[wave][mo + r] = s[r]; }
  }
  __syncthreads();
  if (wave < 4 && l < 16) {   // combine the two key-halves (wn=0 and wn=1)
    float m1 = smax[wave][l], s1 = ssum[wave][l];
    float m2 = smax[wave + 4][l], s2 = ssum[wave + 4][l];
    float M = fmaxf(m1, m2);
    float Sm = s1 * __expf(m1 - M) + s2 * __expf(m2 - M);
    int qi = blockIdx.x * 64 + wave * 16 + l;
    rowmax[(size_t)bh * SEQ + qi] = M;
    rowsum[(size_t)bh * SEQ + qi] = Sm;
  }
}

// ---------------------------------------------------------------------------
// 5) P·V pass: read S, p = exp(s-m)/sum, write normalized attn back, and
//    accumulate O = P·V with WMMA (P as f16 A-fragments, V^T as B-fragments).
// ---------------------------------------------------------------------------
__global__ void pv_kernel(float* __restrict__ S,
                          const float* __restrict__ rowmax,
                          const float* __restrict__ rowsum,
                          const _Float16* __restrict__ vT,
                          _Float16* __restrict__ o) {
  int bh = blockIdx.y;
  int b_ = bh / HEADS, h = bh % HEADS;
  int wave = threadIdx.x >> 5;
  int wm = wave & 3, wd = wave >> 2;
  int q0 = blockIdx.x * 64 + wm * 16;
  int l = threadIdx.x & 31;
  int m_ = l & 15, grp = l >> 4;

  float mrow = rowmax[(size_t)bh * SEQ + q0 + m_];
  float inv  = 1.0f / rowsum[(size_t)bh * SEQ + q0 + m_];
  const _Float16* vb = vT + (size_t)bh * DHEAD * SEQ;
  float* Sbase = S + ((size_t)bh * SEQ + q0) * SEQ;

  v8f acc0 = {}, acc1 = {};
  for (int k0 = 0; k0 < SEQ; k0 += 32) {
    float* sp = Sbase + (size_t)m_ * SEQ + k0 + grp * 8;
    v4f f0 = *(const v4f*)sp;
    v4f f1 = *(const v4f*)(sp + 4);
    v4f f2 = *(const v4f*)(sp + 16);
    v4f f3 = *(const v4f*)(sp + 20);
    v4f p0, p1, p2, p3;
#pragma unroll
    for (int i = 0; i < 4; ++i) {
      p0[i] = __expf(f0[i] - mrow) * inv;
      p1[i] = __expf(f1[i] - mrow) * inv;
      p2[i] = __expf(f2[i] - mrow) * inv;
      p3[i] = __expf(f3[i] - mrow) * inv;
    }
    v16h a;
#pragma unroll
    for (int i = 0; i < 4; ++i) {
      a[i]      = (_Float16)p0[i];
      a[4 + i]  = (_Float16)p1[i];
      a[8 + i]  = (_Float16)p2[i];
      a[12 + i] = (_Float16)p3[i];
    }
    __syncthreads();                 // both d-waves finished reading this S slab
    if (wd == 0) {                   // write normalized attn exactly once
      *(v4f*)sp        = p0;
      *(v4f*)(sp + 4)  = p1;
      *(v4f*)(sp + 16) = p2;
      *(v4f*)(sp + 20) = p3;
    }
    v16h bv0 = frag_bt(vb + (size_t)(wd * 32) * SEQ + k0, SEQ);
    v16h bv1 = frag_bt(vb + (size_t)(wd * 32 + 16) * SEQ + k0, SEQ);
    acc0 = wmma16(a, bv0, acc0);
    acc1 = wmma16(a, bv1, acc1);
  }
#pragma unroll
  for (int t = 0; t < 2; ++t) {
    v8f acc = t ? acc1 : acc0;
#pragma unroll
    for (int r = 0; r < 8; ++r) {
      int qi = q0 + grp * 8 + r;
      int d = wd * 32 + t * 16 + (l & 15);
      o[((size_t)b_ * SEQ + qi) * INNER + h * DHEAD + d] = (_Float16)acc[r];
    }
  }
}

// ---------------------------------------------------------------------------
// 6) Output projection: out = O[4096x768]f16 x w_out[768x768]f16 -> f32
// ---------------------------------------------------------------------------
__global__ void out_gemm(const _Float16* __restrict__ o,
                         const _Float16* __restrict__ woT,
                         float* __restrict__ out) {
  int wave = threadIdx.x >> 5;
  int wm = wave & 3, wn = wave >> 2;
  int row0 = blockIdx.y * 64 + wm * 16;
  int col0 = blockIdx.x * 64 + wn * 32;
  v8f acc0 = {}, acc1 = {};
  for (int kk = 0; kk < INNER; kk += 32) {
    v16h a  = frag_a(o + (size_t)row0 * INNER + kk, INNER);
    v16h b0 = frag_bt(woT + (size_t)col0 * INNER + kk, INNER);
    v16h b1 = frag_bt(woT + (size_t)(col0 + 16) * INNER + kk, INNER);
    acc0 = wmma16(a, b0, acc0);
    acc1 = wmma16(a, b1, acc1);
  }
  int l = threadIdx.x & 31;
  int grp = l >> 4;
#pragma unroll
  for (int t = 0; t < 2; ++t) {
    v8f acc = t ? acc1 : acc0;
    int c = col0 + t * 16 + (l & 15);
#pragma unroll
    for (int r = 0; r < 8; ++r) {
      int row = row0 + grp * 8 + r;
      out[(size_t)row * DIM + c] = acc[r];
    }
  }
}

// ---------------------------------------------------------------------------
extern "C" void kernel_launch(void* const* d_in, const int* in_sizes, int n_in,
                              void* d_out, int out_size, void* d_ws, size_t ws_size,
                              hipStream_t stream) {
  const float*     x      = (const float*)d_in[0];
  const long long* labels = (const long long*)d_in[1];
  const float*     con    = (const float*)d_in[2];
  const float*     gamma  = (const float*)d_in[3];
  const float*     beta   = (const float*)d_in[4];
  const float*     w_qkv  = (const float*)d_in[5];
  const float*     w_out  = (const float*)d_in[6];
  (void)in_sizes; (void)n_in; (void)out_size; (void)ws_size;

  float* out  = (float*)d_out;                       // [b, n, 768]
  float* attn = (float*)d_out + (size_t)NROWS * DIM; // [b, h, n, n]

  char* wsb = (char*)d_ws;
  size_t off = 0;
  _Float16* xn    = (_Float16*)(wsb + off); off += (size_t)NROWS * DIM * 2;
  _Float16* wqkvT = (_Float16*)(wsb + off); off += (size_t)QKV3 * DIM * 2;
  _Float16* woT   = (_Float16*)(wsb + off); off += (size_t)DIM * INNER * 2;
  _Float16* qf    = (_Float16*)(wsb + off); off += (size_t)BATCH * HEADS * SEQ * DHEAD * 2;
  _Float16* kf    = (_Float16*)(wsb + off); off += (size_t)BATCH * HEADS * SEQ * DHEAD * 2;
  _Float16* vTf   = (_Float16*)(wsb + off); off += (size_t)BATCH * HEADS * SEQ * DHEAD * 2;
  _Float16* of    = (_Float16*)(wsb + off); off += (size_t)NROWS * INNER * 2;
  float* rowmax   = (float*)(wsb + off);    off += (size_t)BATCH * HEADS * SEQ * 4;
  float* rowsum   = (float*)(wsb + off);    off += (size_t)BATCH * HEADS * SEQ * 4;

  ln_kernel<<<NROWS, 256, 0, stream>>>(x, gamma, beta, xn);
  wt_kernel<<<(DIM * QKV3 + 255) / 256, 256, 0, stream>>>(w_qkv, wqkvT, DIM, QKV3);
  wt_kernel<<<(INNER * DIM + 255) / 256, 256, 0, stream>>>(w_out, woT, INNER, DIM);
  qkv_gemm<<<dim3(QKV3 / 64, NROWS / 64), 256, 0, stream>>>(xn, wqkvT, qf, kf, vTf);
  s_kernel<<<dim3(SEQ / 64, BATCH * HEADS), 256, 0, stream>>>(
      qf, kf, labels, con, attn, rowmax, rowsum);
  pv_kernel<<<dim3(SEQ / 64, BATCH * HEADS), 256, 0, stream>>>(
      attn, rowmax, rowsum, vTf, of);
  out_gemm<<<dim3(DIM / 64, NROWS / 64), 256, 0, stream>>>(of, woT, out);
}
